// MSA_Outer_Product_Mean_41583873360513
// MI455X (gfx1250) — matrix operand order repaired and checked by reading, not compiled, not run on hardware
//
#include <hip/hip_runtime.h>

#define MSA_DIM  256
#define PAIR_DIM 128
#define HIDDEN   32
#define R_SEQ    128
#define L_SEQ    256
#define EPS_LN   1e-5f

typedef __bf16 bf16_t;
typedef __attribute__((ext_vector_type(16))) __bf16 bf16x16;
typedef __attribute__((ext_vector_type(8)))  __bf16 bf16x8;
typedef __attribute__((ext_vector_type(8)))  float  f32x8;
typedef __attribute__((ext_vector_type(4)))  float  f32x4;

#define SHUF16(lo, hi) __builtin_shufflevector((lo), (hi), 0,1,2,3,4,5,6,7,8,9,10,11,12,13,14,15)

// ---------------------------------------------------------------------------
// Kernel 0: LayerNorm over embed dim + two H=32 projections, emit bf16 in
// WMMA-friendly layouts:
//   m1buf[(r*L + j)*32 + b]  (B-matrix layout for phase 1: lane<->k=r, contiguous n=b)
//   m2T  [(i*32 + a)*128 + r] (A-matrix layout for phase 1: row=(i,a), contiguous k=r)
// One wave per (r,l) row; wave32 shuffle reductions.
// ---------------------------------------------------------------------------
__global__ __launch_bounds__(256) void ln_proj_kernel(
    const float* __restrict__ msa, const float* __restrict__ gamma,
    const float* __restrict__ beta, const float* __restrict__ w1,
    const float* __restrict__ b1,  const float* __restrict__ w2,
    const float* __restrict__ b2,  bf16_t* __restrict__ m1buf,
    bf16_t* __restrict__ m2T)
{
    __shared__ __align__(16) float smem[8][MSA_DIM];
    const int lane = threadIdx.x & 31;
    const int wv   = threadIdx.x >> 5;
    const int row  = blockIdx.x * 8 + wv;      // row = r*L + l
    const int r    = row >> 8;                 // / L_SEQ
    const int l    = row & (L_SEQ - 1);

    const float* rp = msa + (size_t)row * MSA_DIM;
    f32x4 x0 = ((const f32x4*)rp)[lane * 2];
    f32x4 x1 = ((const f32x4*)rp)[lane * 2 + 1];

    float s = x0[0]+x0[1]+x0[2]+x0[3] + x1[0]+x1[1]+x1[2]+x1[3];
    #pragma unroll
    for (int m = 16; m >= 1; m >>= 1) s += __shfl_xor(s, m, 32);
    const float mu = s * (1.0f / MSA_DIM);

    float vs = 0.f;
    #pragma unroll
    for (int t = 0; t < 4; t++) { float d = x0[t]-mu; vs += d*d; }
    #pragma unroll
    for (int t = 0; t < 4; t++) { float d = x1[t]-mu; vs += d*d; }
    #pragma unroll
    for (int m = 16; m >= 1; m >>= 1) vs += __shfl_xor(vs, m, 32);
    const float rstd = rsqrtf(vs * (1.0f / MSA_DIM) + EPS_LN);

    const int e0 = lane * 8;
    #pragma unroll
    for (int t = 0; t < 8; t++) {
        float x = (t < 4) ? x0[t] : x1[t - 4];
        int e = e0 + t;
        smem[wv][e] = (x - mu) * rstd * gamma[e] + beta[e];
    }
    // wave-private LDS row: DS ops are in-order within a wave, no barrier needed

    const int h = lane;                         // each lane owns one hidden unit
    const float* w1r = w1 + (size_t)h * MSA_DIM;
    const float* w2r = w2 + (size_t)h * MSA_DIM;
    float acc1 = 0.f, acc2 = 0.f;
    #pragma unroll 4
    for (int e = 0; e < MSA_DIM; e += 4) {
        f32x4 mv = *(const f32x4*)&smem[wv][e];
        f32x4 a  = *(const f32x4*)(w1r + e);
        f32x4 c  = *(const f32x4*)(w2r + e);
        acc1 += mv[0]*a[0] + mv[1]*a[1] + mv[2]*a[2] + mv[3]*a[3];
        acc2 += mv[0]*c[0] + mv[1]*c[1] + mv[2]*c[2] + mv[3]*c[3];
    }
    m1buf[(size_t)row * HIDDEN + h]                 = (bf16_t)(acc1 + b1[h]);
    m2T[((size_t)l * HIDDEN + h) * R_SEQ + r]       = (bf16_t)(acc2 + b2[h]);
}

// ---------------------------------------------------------------------------
// Kernel 1: w3 (128 x 1024) -> bf16 w3T (1024 x 128), B-matrix layout for
// phase 2 (lane<->k, contiguous n=p).
// ---------------------------------------------------------------------------
__global__ __launch_bounds__(256) void w3t_kernel(
    const float* __restrict__ w3, bf16_t* __restrict__ w3T)
{
    int idx = blockIdx.x * 256 + threadIdx.x;   // idx = k*128 + p
    int k = idx >> 7, p = idx & 127;
    w3T[idx] = (bf16_t)w3[(size_t)p * (HIDDEN * HIDDEN) + k];
}

// ---------------------------------------------------------------------------
// Kernel 2: fused outer-product-mean + pair projection.
// Block = 256 threads (8 waves), owns a 4x4 tile of (i,j) pairs.
// Phase 1: opm[(i,a),(j,b)] = sum_r m2T[(i,a),r] * m1[r,(j,b)]  (K=128, bf16 WMMA)
//          -> scaled by 1/R, written bf16 into padded LDS (16 pairs x 1024).
// Phase 2: out[pair, p] = opm_LDS (16x1024) x w3T (1024x128) + b3  (K=1024).
// ---------------------------------------------------------------------------
__global__ __launch_bounds__(256) void opm_fused_kernel(
    const bf16_t* __restrict__ m1buf,  // (R*L, 32)
    const bf16_t* __restrict__ m2T,    // (L*32, 128)
    const bf16_t* __restrict__ w3T,    // (1024, 128)
    const float*  __restrict__ b3,     // (128)
    float*        __restrict__ out)    // (L, L, 128)
{
    // +8 bf16 pad per row: row stride 2064B => phase-2 lanes (one row each,
    // same k offset) land on distinct LDS bank groups.
    __shared__ __align__(16) bf16_t opm[16][HIDDEN * HIDDEN + 8];

    const int tid  = threadIdx.x;
    const int lane = tid & 31;
    const int wv   = tid >> 5;
    const int h    = lane >> 4;     // K-half select per WMMA A layout
    const int ln15 = lane & 15;
    const int i0   = blockIdx.x * 4;
    const int j0   = blockIdx.y * 4;
    const float invR = 1.0f / R_SEQ;

    // ---- Phase 1: 64 (i_loc, a-half, j_loc, b-half) tiles over 8 waves ----
    for (int t = wv; t < 64; t += 8) {
        const int i_loc = (t >> 4) & 3;
        const int ah    = (t >> 3) & 1;
        const int j_loc = (t >> 1) & 3;
        const int bh    =  t       & 1;
        const int a0 = ah * 16, b0 = bh * 16;

        // A row (fixed (i,a)=m, k=r contiguous)
        const bf16_t* arow = m2T + ((size_t)((i0 + i_loc) * HIDDEN + a0 + ln15)) * R_SEQ;
        f32x8 c = {};
        #pragma unroll
        for (int kc = 0; kc < 4; kc++) {
            const int rc0 = kc * 32;
            bf16x8 alo = *(const bf16x8*)(arow + rc0 + h * 8);
            bf16x8 ahi = *(const bf16x8*)(arow + rc0 + 16 + h * 8);
            bf16x16 afrag = SHUF16(alo, ahi);
            // B col: n = b0 + ln15, k = rc0 + table order (stride L*32)
            const bf16_t* brow = m1buf
                + ((size_t)rc0 * L_SEQ + (j0 + j_loc)) * HIDDEN + b0 + ln15;
            bf16x16 bfrag;
            #pragma unroll
            for (int s = 0; s < 16; s++) {
                const int kk = (s < 8) ? (h * 8 + s) : (16 + h * 8 + (s - 8));
                bfrag[s] = brow[(size_t)kk * L_SEQ * HIDDEN];
            }
            c = __builtin_amdgcn_wmma_f32_16x16x32_bf16(
                    false, afrag, false, bfrag, (short)0, c, false, false);
        }
        const int pair = i_loc * 4 + j_loc;
        #pragma unroll
        for (int rr = 0; rr < 8; rr++) {
            int a = a0 + rr + 8 * h;       // D: VGPR rr -> M = rr (+8 for hi lanes)
            int b = b0 + ln15;             //    lane&15 -> N
            opm[pair][a * HIDDEN + b] = (bf16_t)(c[rr] * invR);
        }
    }
    __syncthreads();

    // ---- Phase 2: projection, one 16-wide p-tile per wave, K = 1024 ----
    const int p0 = wv * 16;
    f32x8 c = {};
    for (int kc = 0; kc < 32; kc++) {
        const bf16_t* ar = &opm[ln15][kc * 32 + h * 8];
        bf16x8 alo = *(const bf16x8*)ar;
        bf16x8 ahi = *(const bf16x8*)(ar + 16);
        bf16x16 afrag = SHUF16(alo, ahi);
        const bf16_t* br = w3T + (size_t)(kc * 32) * PAIR_DIM + p0 + ln15;
        bf16x16 bfrag;
        #pragma unroll
        for (int s = 0; s < 16; s++) {
            const int kk = (s < 8) ? (h * 8 + s) : (16 + h * 8 + (s - 8));
            bfrag[s] = br[(size_t)kk * PAIR_DIM];
        }
        c = __builtin_amdgcn_wmma_f32_16x16x32_bf16(
                false, afrag, false, bfrag, (short)0, c, false, false);
    }
    const float bias = b3[p0 + ln15];
    #pragma unroll
    for (int rr = 0; rr < 8; rr++) {
        int pair = rr + 8 * h;
        int i_g = i0 + (pair >> 2);
        int j_g = j0 + (pair & 3);
        out[((size_t)i_g * L_SEQ + j_g) * PAIR_DIM + p0 + ln15] = c[rr] + bias;
    }
}

// ---------------------------------------------------------------------------
extern "C" void kernel_launch(void* const* d_in, const int* in_sizes, int n_in,
                              void* d_out, int out_size, void* d_ws, size_t ws_size,
                              hipStream_t stream)
{
    (void)in_sizes; (void)n_in; (void)out_size; (void)ws_size;
    const float* msa   = (const float*)d_in[0];
    const float* gamma = (const float*)d_in[1];
    const float* beta  = (const float*)d_in[2];
    const float* w1    = (const float*)d_in[3];
    const float* b1    = (const float*)d_in[4];
    const float* w2    = (const float*)d_in[5];
    const float* b2    = (const float*)d_in[6];
    const float* w3    = (const float*)d_in[7];
    const float* b3    = (const float*)d_in[8];
    float* out = (float*)d_out;

    bf16_t* m1buf = (bf16_t*)d_ws;                               // R*L*32  = 1M bf16 (2 MB)
    bf16_t* m2T   = m1buf + (size_t)R_SEQ * L_SEQ * HIDDEN;      // L*32*R  = 1M bf16 (2 MB)
    bf16_t* w3T   = m2T   + (size_t)L_SEQ * HIDDEN * R_SEQ;      // 1024*128 bf16 (256 KB)

    hipLaunchKernelGGL(ln_proj_kernel, dim3(R_SEQ * L_SEQ / 8), dim3(256), 0, stream,
                       msa, gamma, beta, w1, b1, w2, b2, m1buf, m2T);
    hipLaunchKernelGGL(w3t_kernel, dim3((HIDDEN * HIDDEN * PAIR_DIM) / 256), dim3(256),
                       0, stream, w3, w3T);
    hipLaunchKernelGGL(opm_fused_kernel, dim3(L_SEQ / 4, L_SEQ / 4), dim3(256), 0, stream,
                       m1buf, m2T, w3T, b3, out);
}
